// InputMemorySelfAtt_45990509805721
// MI455X (gfx1250) — matrix operand order repaired and checked by reading, not compiled, not run on hardware
//
#include <hip/hip_runtime.h>
#include <math.h>

// ---------------- model constants ----------------
constexpr int BB = 8, SS = 448, MMEM = 64, HH = 768, NLAYERS = 2, NHEADS = 12;
constexpr int SWL = 512, DHEAD = 64, FFD = 3072, SELD = 512;
constexpr int NROWS = BB * SWL;            // 4096
constexpr float NEGF = -1e9f;

typedef __bf16 bf16;
typedef bf16 v16bf __attribute__((ext_vector_type(16)));
typedef float v8f  __attribute__((ext_vector_type(8)));

union BF16Frag { v16bf v; uint4 q[2]; };

__device__ __forceinline__ v8f wmma_bf16(v16bf a, v16bf b, v8f c) {
  return __builtin_amdgcn_wmma_f32_16x16x32_bf16(false, a, false, b, (short)0, c, false, false);
}

__device__ __forceinline__ void wave_lds_fence() {
  asm volatile("s_wait_dscnt 0" ::: "memory");
  __builtin_amdgcn_wave_barrier();
}

// CDNA5 async global->LDS copy (ASYNCcnt-tracked), 16B per lane.
// lds_off: byte offset within the workgroup LDS allocation (low 32 bits of a
// generic pointer to __shared__, per the LDS aperture rule addr[31:0]).
__device__ __forceinline__ void async_ld_b128(unsigned lds_off, const void* g) {
  asm volatile("global_load_async_to_lds_b128 %0, %1, off"
               :: "v"(lds_off), "v"((unsigned long long)(uintptr_t)g)
               : "memory");
}
__device__ __forceinline__ void wait_async0() {
  asm volatile("s_wait_asynccnt 0x0" ::: "memory");
}

// ---------------- small helpers ----------------
__global__ void cvt_f32_bf16_k(const float* __restrict__ s, bf16* __restrict__ d, int n) {
  int i = blockIdx.x * 256 + threadIdx.x;
  if (i < n) d[i] = (bf16)s[i];
}

__global__ void order_k(const int* __restrict__ mask, int* __restrict__ order, int* __restrict__ Lb) {
  int b = blockIdx.x;
  if (threadIdx.x != 0) return;
  int cnt = 0;
  for (int i = 0; i < SS; ++i) if (mask[b * SS + i]) order[b * SS + cnt++] = i;
  Lb[b] = cnt;
  for (int i = 0; i < SS; ++i) if (!mask[b * SS + i]) order[b * SS + cnt++] = i;
}

__device__ __forceinline__ float block_sum256(float v, float* red) {
  int tid = threadIdx.x;
  red[tid] = v; __syncthreads();
  for (int s = 128; s > 0; s >>= 1) { if (tid < s) red[tid] += red[tid + s]; __syncthreads(); }
  float r = red[0]; __syncthreads();
  return r;
}

// ---------------- embedding + splice + LN ----------------
__global__ __launch_bounds__(256) void embed_ln_k(
    const int* __restrict__ ids, const int* __restrict__ tt,
    const int* __restrict__ order, const int* __restrict__ Lb,
    const float* __restrict__ etab, const float* __restrict__ pos,
    const float* __restrict__ ttemb, const float* __restrict__ g, const float* __restrict__ bt,
    const float* __restrict__ memory,
    float* __restrict__ xf, bf16* __restrict__ xh,
    float* __restrict__ maskneg, int* __restrict__ wttbuf)
{
  __shared__ float red[256];
  const int row = blockIdx.x;
  const int b = row / SWL, j = row % SWL;
  const int tid = threadIdx.x;
  const int L = Lb[b];
  const bool is_mem = (j >= L) && (j < L + MMEM);
  const float* src;
  int wtt;
  if (is_mem) { src = memory + (size_t)(j - L) * HH; wtt = 1; }
  else {
    int tok = (j < L) ? j : (j - MMEM);
    tok = tok < 0 ? 0 : (tok > SS - 1 ? SS - 1 : tok);
    int orig = order[b * SS + tok];
    src = etab + (size_t)ids[b * SS + orig] * HH;
    wtt = (j < L) ? tt[b * SS + orig] : 1;
  }
  float v[3];
  for (int e = 0; e < 3; ++e) {
    int i = tid + e * 256;
    v[e] = src[i] + pos[(size_t)j * HH + i] + ttemb[(size_t)wtt * HH + i];
  }
  float mean = block_sum256(v[0] + v[1] + v[2], red) / (float)HH;
  float d0 = v[0] - mean, d1 = v[1] - mean, d2 = v[2] - mean;
  float var = block_sum256(d0 * d0 + d1 * d1 + d2 * d2, red) / (float)HH;
  float inv = rsqrtf(var + 1e-12f);
  for (int e = 0; e < 3; ++e) {
    int i = tid + e * 256;
    float y = (v[e] - mean) * inv * g[i] + bt[i];
    xf[(size_t)row * HH + i] = y;
    xh[(size_t)row * HH + i] = (bf16)y;
  }
  if (tid == 0) {
    maskneg[row] = (j < L + MMEM) ? 0.f : NEGF;
    wttbuf[row] = wtt;
  }
}

// ---------------- residual + LN (in-place on x) ----------------
__global__ __launch_bounds__(256) void resid_ln_k(
    float* __restrict__ xf, bf16* __restrict__ xh, const float* __restrict__ add,
    const float* __restrict__ g, const float* __restrict__ bt)
{
  __shared__ float red[256];
  const int row = blockIdx.x, tid = threadIdx.x;
  float v[3];
  for (int e = 0; e < 3; ++e) {
    int i = tid + e * 256;
    v[e] = xf[(size_t)row * HH + i] + add[(size_t)row * HH + i];
  }
  float mean = block_sum256(v[0] + v[1] + v[2], red) / (float)HH;
  float d0 = v[0] - mean, d1 = v[1] - mean, d2 = v[2] - mean;
  float var = block_sum256(d0 * d0 + d1 * d1 + d2 * d2, red) / (float)HH;
  float inv = rsqrtf(var + 1e-12f);
  for (int e = 0; e < 3; ++e) {
    int i = tid + e * 256;
    float y = (v[e] - mean) * inv * g[i] + bt[i];
    xf[(size_t)row * HH + i] = y;
    xh[(size_t)row * HH + i] = (bf16)y;
  }
}

// ---------------- WMMA GEMM: C[MxN] = A[MxK](bf16) * W[KxN](bf16) + bias ----------------
template<int ACT> __device__ __forceinline__ float act_apply(float x) {
  if (ACT == 1) return fmaxf(x, 0.f);
  if (ACT == 2) { float x3 = x * x * x; return 0.5f * x * (1.f + tanhf(0.7978845608028654f * (x + 0.044715f * x3))); }
  if (ACT == 3) return tanhf(x);
  if (ACT == 4) return 1.f / (1.f + __expf(-x));
  return x;
}

template<int ACT>
__global__ __launch_bounds__(256) void gemm_bf16_k(
    const bf16* __restrict__ A, const bf16* __restrict__ W,
    const float* __restrict__ bias, float* __restrict__ outF, bf16* __restrict__ outH,
    int M, int N, int K)
{
  __shared__ bf16 As[2][128 * 32];   // [row][k], double-buffered (async-filled)
  __shared__ bf16 Bs[2][64 * 40];    // [col][k] transposed, pitch 40, double-buffered
  const int tid = threadIdx.x, lane = tid & 31, wave = tid >> 5;
  const int wm = wave >> 1, wn = wave & 1;
  const int mBase = blockIdx.y * 128, nBase = blockIdx.x * 64;
  const int half = lane >> 4, l16 = lane & 15;
  const int abase = half * 8, kb16 = half * 16;

  // per-thread A-tile chunks: idx = tid + c*256 -> (row, 8-elem segment)
  const int r0 = (tid) >> 2, sg0 = (tid) & 3;
  const int r1 = (tid + 256) >> 2, sg1 = (tid + 256) & 3;
  const unsigned asb[2] = { (unsigned)(uintptr_t)&As[0][0], (unsigned)(uintptr_t)&As[1][0] };
  // per-thread B-tile chunk: one row-segment of 8 elems
  const int bk = tid >> 3, bseg = tid & 7;

  v8f acc[2][2];
  for (int a = 0; a < 2; ++a) for (int b = 0; b < 2; ++b) for (int v = 0; v < 8; ++v) acc[a][b][v] = 0.f;

  const int nk = K >> 5;
  // ---- prologue: async-fill A(0), fetch B(0) into regs ----
  async_ld_b128(asb[0] + (unsigned)(r0 * 32 + sg0 * 8) * 2,
                &A[(size_t)(mBase + r0) * K + sg0 * 8]);
  async_ld_b128(asb[0] + (unsigned)(r1 * 32 + sg1 * 8) * 2,
                &A[(size_t)(mBase + r1) * K + sg1 * 8]);
  uint4 wv = *(const uint4*)&W[(size_t)bk * N + nBase + bseg * 8];

  for (int i = 0; i < nk; ++i) {
    const int cur = i & 1;
    // transpose-store B(i) regs into Bs[cur]
    {
      const bf16* wp = reinterpret_cast<const bf16*>(&wv);
      for (int e = 0; e < 8; ++e) Bs[cur][(bseg * 8 + e) * 40 + bk] = wp[e];
    }
    wait_async0();           // A(i) resident in LDS (per-wave ASYNCcnt)
    __syncthreads();         // B(i) visible block-wide
    if (i + 1 < nk) {
      const int k1 = (i + 1) << 5;
      async_ld_b128(asb[cur ^ 1] + (unsigned)(r0 * 32 + sg0 * 8) * 2,
                    &A[(size_t)(mBase + r0) * K + k1 + sg0 * 8]);
      async_ld_b128(asb[cur ^ 1] + (unsigned)(r1 * 32 + sg1 * 8) * 2,
                    &A[(size_t)(mBase + r1) * K + k1 + sg1 * 8]);
      wv = *(const uint4*)&W[(size_t)(k1 + bk) * N + nBase + bseg * 8];
      if (i + 2 < nk)        // prefetch W two tiles ahead into L2 (global_prefetch_b8)
        __builtin_prefetch(&W[(size_t)(((i + 2) << 5) + bk) * N + nBase + bseg * 8], 0, 1);
    }
    BF16Frag af[2], bfr[2];
    for (int mt = 0; mt < 2; ++mt) {
      int r = wm * 32 + mt * 16 + l16;
      af[mt].q[0] = *(const uint4*)&As[cur][r * 32 + kb16];
      af[mt].q[1] = *(const uint4*)&As[cur][r * 32 + kb16 + 8];
    }
    for (int nt = 0; nt < 2; ++nt) {
      int n = wn * 32 + nt * 16 + l16;
      bfr[nt].q[0] = *(const uint4*)&Bs[cur][n * 40 + kb16];
      bfr[nt].q[1] = *(const uint4*)&Bs[cur][n * 40 + kb16 + 8];
    }
    for (int mt = 0; mt < 2; ++mt)
      for (int nt = 0; nt < 2; ++nt)
        acc[mt][nt] = wmma_bf16(af[mt].v, bfr[nt].v, acc[mt][nt]);
    __syncthreads();         // all waves done reading buffer `cur` before refill at i+2
  }
  for (int mt = 0; mt < 2; ++mt)
    for (int nt = 0; nt < 2; ++nt) {
      int col = nBase + wn * 32 + nt * 16 + l16;
      float bv = bias ? bias[col] : 0.f;
      for (int v = 0; v < 8; ++v) {
        int row = mBase + wm * 32 + mt * 16 + half * 8 + v;
        float x = act_apply<ACT>(acc[mt][nt][v] + bv);
        size_t o = (size_t)row * N + col;
        if (outF) outF[o] = x;
        if (outH) outH[o] = (bf16)x;
      }
    }
}

// ---------------- QKV split f32 -> bf16 head-major ----------------
__global__ void qkv_split_k(const float* __restrict__ qkv,
                            bf16* __restrict__ Qh, bf16* __restrict__ Kh, bf16* __restrict__ Vh)
{
  int gid = blockIdx.x * 256 + threadIdx.x;       // NROWS * 2304 total
  int srow = gid / (3 * HH), c = gid % (3 * HH);
  int b = srow / SWL, s = srow % SWL;
  int which = c / HH, hc = c % HH;
  int h = hc / DHEAD, d = hc % DHEAD;
  float v = qkv[(size_t)gid];
  size_t dst = (((size_t)b * NHEADS + h) * SWL + s) * DHEAD + d;
  if (which == 0) Qh[dst] = (bf16)(v * 0.125f);   // 1/sqrt(64)
  else if (which == 1) Kh[dst] = (bf16)v;
  else Vh[dst] = (bf16)v;
}

// ---------------- flash-style attention: one block per (b,h) ----------------
__global__ __launch_bounds__(256) void attention_k(
    const bf16* __restrict__ Qh, const bf16* __restrict__ Kh, const bf16* __restrict__ Vh,
    const float* __restrict__ maskneg, bf16* __restrict__ ctx)
{
  extern __shared__ char smem[];
  const int bh = blockIdx.x, b = bh / NHEADS, h = bh % NHEADS;
  bf16* vt   = (bf16*)smem;                              // [64][520]
  float* scA = (float*)(smem + 64 * 520 * 2);            // 8 waves x 16x32
  bf16*  paA = (bf16*)((char*)scA + 8 * 16 * 32 * 4);    // 8 waves x 16x32
  float* frA = (float*)((char*)paA + 8 * 16 * 32 * 2);   // 8 waves x 16
  const int tid = threadIdx.x, lane = tid & 31, wave = tid >> 5;
  float* sc = scA + wave * (16 * 32);
  bf16*  pa = paA + wave * (16 * 32);
  float* fr = frA + wave * 16;
  const bf16* Qb = Qh + (size_t)bh * SWL * DHEAD;
  const bf16* Kb = Kh + (size_t)bh * SWL * DHEAD;
  const bf16* Vb = Vh + (size_t)bh * SWL * DHEAD;
  const float* mrow = maskneg + b * SWL;

  for (int i = tid; i < SWL * DHEAD; i += 256) {
    int s = i >> 6, d = i & 63;
    vt[d * 520 + s] = Vb[i];
  }
  __syncthreads();

  const int half = lane >> 4, l16 = lane & 15;
  const int abase = half * 8, kb16 = half * 16;

  for (int qi = 0; qi < 4; ++qi) {
    const int q0 = (wave * 4 + qi) * 16;
    BF16Frag qa[2];
    {
      const bf16* qp = Qb + (size_t)(q0 + l16) * DHEAD;
      qa[0].q[0] = *(const uint4*)&qp[kb16];
      qa[0].q[1] = *(const uint4*)&qp[kb16 + 8];
      qa[1].q[0] = *(const uint4*)&qp[32 + kb16];
      qa[1].q[1] = *(const uint4*)&qp[32 + kb16 + 8];
    }
    v8f occ[4];
    for (int dc = 0; dc < 4; ++dc) for (int v = 0; v < 8; ++v) occ[dc][v] = 0.f;
    float mstat = -3e38f, sstat = 0.f;   // valid in lanes < 16

    for (int kp = 0; kp < 16; ++kp) {
      for (int s2 = 0; s2 < 2; ++s2) {
        int key = (kp * 2 + s2) * 16 + l16;
        const bf16* kpnt = Kb + (size_t)key * DHEAD;
        BF16Frag kf0, kf1;
        kf0.q[0] = *(const uint4*)&kpnt[kb16];
        kf0.q[1] = *(const uint4*)&kpnt[kb16 + 8];
        kf1.q[0] = *(const uint4*)&kpnt[32 + kb16];
        kf1.q[1] = *(const uint4*)&kpnt[32 + kb16 + 8];
        v8f s8; for (int v = 0; v < 8; ++v) s8[v] = 0.f;
        s8 = wmma_bf16(qa[0].v, kf0.v, s8);
        s8 = wmma_bf16(qa[1].v, kf1.v, s8);
        float biasc = mrow[key];
        for (int v = 0; v < 8; ++v)
          sc[(half * 8 + v) * 32 + s2 * 16 + l16] = s8[v] + biasc;
      }
      wave_lds_fence();
      if (lane < 16) {
        float tmax = -3e38f;
        for (int j = 0; j < 32; ++j) tmax = fmaxf(tmax, sc[lane * 32 + j]);
        float mn = fmaxf(mstat, tmax);
        float f = __expf(mstat - mn);
        float ssum = 0.f;
        for (int j = 0; j < 32; ++j) {
          float p = __expf(sc[lane * 32 + j] - mn);
          ssum += p;
          pa[lane * 32 + j] = (bf16)p;
        }
        sstat = sstat * f + ssum;
        mstat = mn;
        fr[lane] = f;
      }
      wave_lds_fence();
      float fv[8];
      for (int v = 0; v < 8; ++v) fv[v] = fr[half * 8 + v];
      for (int dc = 0; dc < 4; ++dc)
        for (int v = 0; v < 8; ++v) occ[dc][v] *= fv[v];
      BF16Frag paf;
      paf.q[0] = *(const uint4*)&pa[l16 * 32 + kb16];
      paf.q[1] = *(const uint4*)&pa[l16 * 32 + kb16 + 8];
      for (int dc = 0; dc < 4; ++dc) {
        int d = dc * 16 + l16;
        BF16Frag vf;
        vf.q[0] = *(const uint4*)&vt[d * 520 + kp * 32 + kb16];
        vf.q[1] = *(const uint4*)&vt[d * 520 + kp * 32 + kb16 + 8];
        occ[dc] = wmma_bf16(paf.v, vf.v, occ[dc]);
      }
    }
    if (lane < 16) fr[lane] = 1.f / sstat;
    wave_lds_fence();
    float iv[8];
    for (int v = 0; v < 8; ++v) iv[v] = fr[half * 8 + v];
    for (int dc = 0; dc < 4; ++dc)
      for (int v = 0; v < 8; ++v) {
        int row = q0 + half * 8 + v;
        int d = dc * 16 + l16;
        ctx[((size_t)b * SWL + row) * HH + h * DHEAD + d] = (bf16)(occ[dc][v] * iv[v]);
      }
  }
}

// ---------------- attention-pool head: sigmoid(a1 . w2 + b2) per row ----------------
__global__ __launch_bounds__(256) void att2_k(const bf16* __restrict__ a1,
                                              const float* __restrict__ w, const float* __restrict__ b2,
                                              float* __restrict__ wvec)
{
  __shared__ float red[256];
  const int row = blockIdx.x, tid = threadIdx.x;
  float s = 0.f;
  for (int i = tid; i < 2 * HH; i += 256)
    s += (float)a1[(size_t)row * (2 * HH) + i] * w[i];
  float tot = block_sum256(s, red);
  if (tid == 0) wvec[row] = 1.f / (1.f + __expf(-(tot + b2[0])));
}

// ---------------- masked-softmax pooling -> rep[b][SEL] ----------------
__global__ __launch_bounds__(512) void pool_k(const float* __restrict__ wvec, const int* __restrict__ wtt,
                                              const float* __restrict__ val, float* __restrict__ rep)
{
  __shared__ float fw[512];
  __shared__ float red[512];
  const int b = blockIdx.x, tid = threadIdx.x;
  float logit = wvec[b * SWL + tid] + (wtt[b * SWL + tid] == 1 ? NEGF : 0.f);
  red[tid] = logit; __syncthreads();
  for (int s = 256; s > 0; s >>= 1) { if (tid < s) red[tid] = fmaxf(red[tid], red[tid + s]); __syncthreads(); }
  float mx = red[0]; __syncthreads();
  float e = __expf(logit - mx);
  red[tid] = e; __syncthreads();
  for (int s = 256; s > 0; s >>= 1) { if (tid < s) red[tid] += red[tid + s]; __syncthreads(); }
  float denom = red[0]; __syncthreads();
  fw[tid] = e / denom;
  __syncthreads();
  float acc = 0.f;
  for (int s = 0; s < SWL; ++s)
    acc += fw[s] * val[((size_t)b * SWL + s) * SELD + tid];
  rep[b * SELD + tid] = fmaxf(acc, 0.f);
}

// ---------------- classifier ----------------
__global__ void cat_k(const float* __restrict__ reps, float* __restrict__ xcat) {
  int i = blockIdx.x * 256 + threadIdx.x;
  if (i >= BB * 3 * SELD) return;
  int b = i / (3 * SELD), c = i % (3 * SELD);
  xcat[i] = reps[(c / SELD) * BB * SELD + b * SELD + (c % SELD)];
}

__global__ __launch_bounds__(256) void mlp_k(const float* __restrict__ in, const float* __restrict__ w,
                                             const float* __restrict__ bias,
                                             const float* __restrict__ g, const float* __restrict__ bt,
                                             float* __restrict__ out, int K, int N, int bnrelu)
{
  const int b = blockIdx.x;
  const float bnscale = rsqrtf(1.f + 1e-5f);
  for (int j = threadIdx.x; j < N; j += 256) {
    float acc = bias[j];
    for (int k = 0; k < K; ++k) acc += in[(size_t)b * K + k] * w[(size_t)k * N + j];
    if (bnrelu) { acc = fmaxf(acc, 0.f); acc = acc * bnscale * g[j] + bt[j]; }
    out[(size_t)b * N + j] = acc;
  }
}

// ---------------- host orchestration ----------------
static void launch_gemm(int act, const bf16* A, const bf16* W, const float* bias,
                        float* oF, bf16* oH, int M, int N, int K, hipStream_t st) {
  dim3 g(N / 64, M / 128), blk(256);
  switch (act) {
    case 0: gemm_bf16_k<0><<<g, blk, 0, st>>>(A, W, bias, oF, oH, M, N, K); break;
    case 1: gemm_bf16_k<1><<<g, blk, 0, st>>>(A, W, bias, oF, oH, M, N, K); break;
    case 2: gemm_bf16_k<2><<<g, blk, 0, st>>>(A, W, bias, oF, oH, M, N, K); break;
    case 3: gemm_bf16_k<3><<<g, blk, 0, st>>>(A, W, bias, oF, oH, M, N, K); break;
  }
}

static void cvt(const float* s, bf16* d, int n, hipStream_t st) {
  cvt_f32_bf16_k<<<(n + 255) / 256, 256, 0, st>>>(s, d, n);
}

extern "C" void kernel_launch(void* const* d_in, const int* in_sizes, int n_in,
                              void* d_out, int out_size, void* d_ws, size_t ws_size,
                              hipStream_t stream) {
  (void)in_sizes; (void)n_in; (void)out_size; (void)ws_size;
  // ---- inputs ----
  const int* ids_all = (const int*)d_in[0];
  const int* tt_all  = (const int*)d_in[1];
  const int* am_all  = (const int*)d_in[2];
  const float* etab   = (const float*)d_in[3];
  const float* posemb = (const float*)d_in[4];
  const float* ttemb  = (const float*)d_in[5];
  const float* elng   = (const float*)d_in[6];
  const float* elnb   = (const float*)d_in[7];
  const float* qkv_w  = (const float*)d_in[8];
  const float* qkv_b  = (const float*)d_in[9];
  const float* o_w    = (const float*)d_in[10];
  const float* o_b    = (const float*)d_in[11];
  const float* ln1g   = (const float*)d_in[12];
  const float* ln1b   = (const float*)d_in[13];
  const float* f1_w   = (const float*)d_in[14];
  const float* f1_b   = (const float*)d_in[15];
  const float* f2_w   = (const float*)d_in[16];
  const float* f2_b   = (const float*)d_in[17];
  const float* ln2g   = (const float*)d_in[18];
  const float* ln2b   = (const float*)d_in[19];
  const float* mem_q  = (const float*)d_in[20];
  const float* mem_a  = (const float*)d_in[21];
  const float* v1_w   = (const float*)d_in[22];
  const float* v1_b   = (const float*)d_in[23];
  const float* v2_w   = (const float*)d_in[24];
  const float* v2_b   = (const float*)d_in[25];
  const float* a1_w   = (const float*)d_in[26];
  const float* a1_b   = (const float*)d_in[27];
  const float* a2_w   = (const float*)d_in[28];
  const float* a2_b   = (const float*)d_in[29];
  const float* c_w1 = (const float*)d_in[30]; const float* c_b1 = (const float*)d_in[31];
  const float* bn1g = (const float*)d_in[32]; const float* bn1b = (const float*)d_in[33];
  const float* c_w2 = (const float*)d_in[34]; const float* c_b2 = (const float*)d_in[35];
  const float* bn2g = (const float*)d_in[36]; const float* bn2b = (const float*)d_in[37];
  const float* c_w3 = (const float*)d_in[38]; const float* c_b3 = (const float*)d_in[39];
  const float* bn3g = (const float*)d_in[40]; const float* bn3b = (const float*)d_in[41];
  const float* c_w4 = (const float*)d_in[42]; const float* c_b4 = (const float*)d_in[43];

  // ---- workspace layout ----
  char* ws = (char*)d_ws;
  size_t cur = 0;
  auto grab = [&](size_t bytes) { void* p = ws + cur; cur = (cur + bytes + 255) & ~(size_t)255; return p; };
  const int nqkvw = NLAYERS * HH * 3 * HH, now = NLAYERS * HH * HH;
  const int nf1w = NLAYERS * HH * FFD, nf2w = NLAYERS * FFD * HH;
  const int nv1w = HH * 2 * SELD, nv2w = 2 * SELD * SELD, na1w = HH * 2 * HH;
  bf16* wqkv_h = (bf16*)grab((size_t)nqkvw * 2);
  bf16* wo_h   = (bf16*)grab((size_t)now * 2);
  bf16* wf1_h  = (bf16*)grab((size_t)nf1w * 2);
  bf16* wf2_h  = (bf16*)grab((size_t)nf2w * 2);
  bf16* wv1_h  = (bf16*)grab((size_t)nv1w * 2);
  bf16* wv2_h  = (bf16*)grab((size_t)nv2w * 2);
  bf16* wa1_h  = (bf16*)grab((size_t)na1w * 2);
  int* order   = (int*)grab((size_t)BB * SS * 4);
  int* Lb      = (int*)grab(BB * 4);
  float* maskneg = (float*)grab((size_t)BB * SWL * 4);
  int*   wttbuf  = (int*)grab((size_t)BB * SWL * 4);
  float* xf    = (float*)grab((size_t)NROWS * HH * 4);
  bf16*  xh    = (bf16*)grab((size_t)NROWS * HH * 2);
  char*  BIG   = (char*)grab((size_t)NROWS * 3 * HH * 4);   // 37.7 MB, aliased
  bf16*  Qh    = (bf16*)grab((size_t)NROWS * HH * 2);
  bf16*  Kh    = (bf16*)grab((size_t)NROWS * HH * 2);
  bf16*  Vh    = (bf16*)grab((size_t)NROWS * HH * 2);
  bf16*  ctxh  = (bf16*)grab((size_t)NROWS * HH * 2);
  float* tmpf  = (float*)grab((size_t)NROWS * HH * 4);
  float* wvec  = (float*)grab((size_t)NROWS * 4);
  float* reps  = (float*)grab((size_t)3 * BB * SELD * 4);
  float* xcat  = (float*)grab((size_t)BB * 3 * SELD * 4);
  float* c1    = (float*)grab((size_t)BB * 3 * SELD * 4);
  float* c2    = (float*)grab((size_t)BB * SELD * 4);
  float* c3    = (float*)grab((size_t)BB * SELD * 4);
  // aliases inside BIG
  float* qkvf = (float*)BIG;                                      // [4096][2304] f32
  bf16*  ffh  = (bf16*)BIG;                                       // [4096][3072] bf16
  bf16*  vh1  = (bf16*)BIG;                                       // [4096][1024] bf16
  bf16*  ah1  = (bf16*)(BIG + (size_t)NROWS * 1024 * 2);          // [4096][1536] bf16
  float* valf = (float*)(BIG + (size_t)NROWS * 1024 * 2 + (size_t)NROWS * 1536 * 2); // [4096][512] f32

  // ---- weight conversion ----
  cvt(qkv_w, wqkv_h, nqkvw, stream);
  cvt(o_w,   wo_h,   now,   stream);
  cvt(f1_w,  wf1_h,  nf1w,  stream);
  cvt(f2_w,  wf2_h,  nf2w,  stream);
  cvt(v1_w,  wv1_h,  nv1w,  stream);
  cvt(v2_w,  wv2_h,  nv2w,  stream);
  cvt(a1_w,  wa1_h,  na1w,  stream);

  const int ATT_SMEM = 64 * 520 * 2 + 8 * 16 * 32 * 4 + 8 * 16 * 32 * 2 + 8 * 16 * 4;

  for (int pass = 0; pass < 3; ++pass) {
    const int* ids = ids_all + pass * BB * SS;
    const int* tt  = tt_all  + pass * BB * SS;
    const int* am  = am_all  + pass * BB * SS;
    const float* memory = (pass == 0) ? mem_q : mem_a;

    order_k<<<BB, 32, 0, stream>>>(am, order, Lb);
    embed_ln_k<<<NROWS, 256, 0, stream>>>(ids, tt, order, Lb, etab, posemb, ttemb,
                                          elng, elnb, memory, xf, xh, maskneg, wttbuf);
    for (int l = 0; l < NLAYERS; ++l) {
      launch_gemm(0, xh, wqkv_h + (size_t)l * HH * 3 * HH, qkv_b + (size_t)l * 3 * HH,
                  qkvf, nullptr, NROWS, 3 * HH, HH, stream);
      qkv_split_k<<<(NROWS * 3 * HH) / 256, 256, 0, stream>>>(qkvf, Qh, Kh, Vh);
      attention_k<<<BB * NHEADS, 256, ATT_SMEM, stream>>>(Qh, Kh, Vh, maskneg, ctxh);
      launch_gemm(0, ctxh, wo_h + (size_t)l * HH * HH, o_b + (size_t)l * HH,
                  tmpf, nullptr, NROWS, HH, HH, stream);
      resid_ln_k<<<NROWS, 256, 0, stream>>>(xf, xh, tmpf, ln1g + (size_t)l * HH, ln1b + (size_t)l * HH);
      launch_gemm(2, xh, wf1_h + (size_t)l * HH * FFD, f1_b + (size_t)l * FFD,
                  nullptr, ffh, NROWS, FFD, HH, stream);
      launch_gemm(0, ffh, wf2_h + (size_t)l * FFD * HH, f2_b + (size_t)l * HH,
                  tmpf, nullptr, NROWS, HH, FFD, stream);
      resid_ln_k<<<NROWS, 256, 0, stream>>>(xf, xh, tmpf, ln2g + (size_t)l * HH, ln2b + (size_t)l * HH);
    }
    // heads
    launch_gemm(1, xh, wv1_h, v1_b, nullptr, vh1, NROWS, 2 * SELD, HH, stream);
    launch_gemm(3, vh1, wv2_h, v2_b, valf, nullptr, NROWS, SELD, 2 * SELD, stream);
    launch_gemm(1, xh, wa1_h, a1_b, nullptr, ah1, NROWS, 2 * HH, HH, stream);
    att2_k<<<NROWS, 256, 0, stream>>>(ah1, a2_w, a2_b, wvec);
    pool_k<<<BB, 512, 0, stream>>>(wvec, wttbuf, valf, reps + (size_t)pass * BB * SELD);
  }

  // classifier
  cat_k<<<(BB * 3 * SELD + 255) / 256, 256, 0, stream>>>(reps, xcat);
  mlp_k<<<BB, 256, 0, stream>>>(xcat, c_w1, c_b1, bn1g, bn1b, c1, 3 * SELD, 3 * SELD, 1);
  mlp_k<<<BB, 256, 0, stream>>>(c1,   c_w2, c_b2, bn2g, bn2b, c2, 3 * SELD, SELD, 1);
  mlp_k<<<BB, 256, 0, stream>>>(c2,   c_w3, c_b3, bn3g, bn3b, c3, SELD, SELD, 1);
  mlp_k<<<BB, 256, 0, stream>>>(c3,   c_w4, c_b4, nullptr, nullptr, (float*)d_out, SELD, 4, 0);
}
